// AdditiveAttention_55018531062333
// MI455X (gfx1250) — compile-verified, matrix operand order checked by reading, not run on hardware
//
#include <hip/hip_runtime.h>
#include <hip/hip_bf16.h>
#include <math.h>

// Problem constants (match reference)
#define BB 64
#define SS 2048
#define QDQ 1024
#define KDK 1024
#define VDV 1024
#define HH 1024

// LDS row stride for staged keys (bytes). 2128/4 = 532 dwords; 532 mod 64 = 20,
// and 20*r mod 64 is distinct for r=0..15 -> conflict-free A-fragment reads.
#define ROWB 2128
#define REG_LDS (16 * ROWB)           // 34,048 B per 16-row region
#define SMEM_BYTES (4 * REG_LDS)      // 136,192 B per WG -> 2 WGs per 320KB WGP
#define SCH 8                         // context-kernel S chunks

typedef __attribute__((ext_vector_type(16))) __bf16 v16bf;
typedef __attribute__((ext_vector_type(8)))  float  v8f;

union Frag { unsigned u[8]; v16bf v; };

__device__ __forceinline__ unsigned bfpack2(float a, float b) {
  unsigned ua = __float_as_uint(a), ub = __float_as_uint(b);
  ua = (ua + 0x7FFFu + ((ua >> 16) & 1u)) >> 16;   // RNE fp32 -> bf16
  ub = (ub + 0x7FFFu + ((ub >> 16) & 1u)) >> 16;
  return ua | (ub << 16);
}

__device__ __forceinline__ float fast_tanh(float x) {
#if __has_builtin(__builtin_amdgcn_tanhf)
  return __builtin_amdgcn_tanhf(x);              // gfx1250 v_tanh_f32
#else
  // tanh(x) = 1 - 2/(exp(2x)+1); v_exp_f32 + v_rcp_f32, no IEEE divide chain
  return 1.0f - 2.0f * __builtin_amdgcn_rcpf(__expf(2.0f * x) + 1.0f);
#endif
}

// ---------------------------------------------------------------------------
// Kernel 1: pack Wk[h][k] (fp32) into bf16 WMMA B-fragment tile order.
// Tile t = ht*32 + kt (ht: 16 h-cols, kt: 32 k-rows). Within a tile, lane l
// owns elements [l*16, l*16+16): N = l&15, K per the 16-bit fragment map
// (VGPR d<4: K=2d+low, d>=4: K=16+2(d-4)+low, +8 for lanes>=16).
// ---------------------------------------------------------------------------
__global__ void pack_wk_bf16(const float* __restrict__ Wk,
                             unsigned short* __restrict__ wkbf) {
  int gid  = blockIdx.x * 256 + threadIdx.x;       // 0 .. H*KD-1
  int t    = gid >> 9;                             // tile index
  int w    = gid & 511;
  int ht   = t >> 5;
  int kt   = t & 31;
  int lane = w >> 4;
  int e    = w & 15;
  int half = lane >> 4;
  int d    = e >> 1;
  int low  = e & 1;
  int kk   = (d < 4 ? 2 * d : 16 + 2 * (d - 4)) + low + half * 8;
  int k    = kt * 32 + kk;
  int n    = ht * 16 + (lane & 15);
  float f  = Wk[(size_t)n * KDK + k];
  unsigned u = __float_as_uint(f);
  u = (u + 0x7FFFu + ((u >> 16) & 1u)) >> 16;
  wkbf[gid] = (unsigned short)u;
}

// ---------------------------------------------------------------------------
// Kernel 2: q_proj[b][h] = dot(query[b], Wq[h])  (tiny: 0.13 GFLOP)
// ---------------------------------------------------------------------------
__global__ void qproj_kernel(const float* __restrict__ query,
                             const float* __restrict__ Wq,
                             float* __restrict__ qproj) {
  int gid = blockIdx.x * 256 + threadIdx.x;        // b*H + h
  int b = gid >> 10;
  int h = gid & 1023;
  const float4* q  = (const float4*)(query + (size_t)b * QDQ);
  const float4* wq = (const float4*)(Wq + (size_t)h * QDQ);
  float acc = 0.0f;
  for (int i = 0; i < QDQ / 4; ++i) {
    float4 a = q[i], c = wq[i];
    acc += a.x * c.x + a.y * c.y + a.z * c.z + a.w * c.w;
  }
  qproj[gid] = acc;
}

// ---------------------------------------------------------------------------
// Kernel 3 (main, WMMA): fused k_proj GEMM + tanh + We-reduction -> scores.
// Grid: B * (S/64) WGs of 8 waves; 136 KB LDS -> 2 WGs/WGP (4 waves/SIMD).
// Region r (16 s-rows) is staged once as bf16 and shared by wave pair
// {r, r+4}; the pair splits the 1024 h-columns (32 h-tiles each).
//  - per h-tile: 32x V_WMMA_F32_16X16X32_BF16 over K=1024
//    A-frag: 2x ds_load_b128 ; B-frag: 2x global_load_b128 (L2-resident Wk)
//  - epilogue: v_tanh_f32(kproj + qproj[h]) * We[h], accumulate per s-row
//  - shfl_xor(1,2,4,8) h-reduction within 16-lane N group, then wave-pair
//    combine through a small static-LDS partial buffer.
// ---------------------------------------------------------------------------
__global__ void fused_scores_kernel(const float* __restrict__ keys,
                                    const float* __restrict__ qproj,
                                    const float* __restrict__ We,
                                    const unsigned short* __restrict__ wkbf,
                                    float* __restrict__ scores) {
  extern __shared__ unsigned char smem[];
  __shared__ float partial[8][16];
  const int b     = blockIdx.x >> 5;        // 32 s-tiles per batch
  const int stile = blockIdx.x & 31;
  const int wave  = threadIdx.x >> 5;
  const int lane  = threadIdx.x & 31;
  const int reg   = wave & 3;               // 16-row staging region
  const int hhalf = wave >> 2;              // which 512-wide h half
  const int s0    = stile * 64 + reg * 16;
  unsigned char* my = smem + reg * REG_LDS;

  // ---- stage: the wave pair splits the 16 rows (8 each), fp32 -> bf16 ----
  for (int r = hhalf * 8; r < hhalf * 8 + 8; ++r) {
    const float* grow = keys + ((size_t)b * SS + (s0 + r)) * KDK;
    for (int c = lane * 4; c < KDK; c += 128) {
      float4 f = *(const float4*)(grow + c);
      uint2 p;
      p.x = bfpack2(f.x, f.y);
      p.y = bfpack2(f.z, f.w);
      *(uint2*)(my + r * ROWB + c * 2) = p;
    }
  }
  __syncthreads();

  float rowacc[8];
#pragma unroll
  for (int r = 0; r < 8; ++r) rowacc[r] = 0.0f;

  const int row  = lane & 15;   // A: M index source ; B/C/D: N (h) index
  const int half = lane >> 4;

  const int ht0 = hhalf * 32;
  for (int ht = ht0; ht < ht0 + 32; ++ht) {
    v8f c = {};
    const unsigned short* bt = wkbf + (size_t)(ht * 32) * 512 + lane * 16;
    __builtin_prefetch(bt, 0, 1);   // global_prefetch_b8 on the Wk stream
#pragma unroll 4
    for (int kt = 0; kt < 32; ++kt) {
      // A fragment from LDS (two b128 loads, conflict-free via ROWB padding)
      const unsigned char* ap = my + row * ROWB + kt * 64 + half * 16;
      uint4 a0 = *(const uint4*)(ap);
      uint4 a1 = *(const uint4*)(ap + 32);
      Frag fa;
      fa.u[0] = a0.x; fa.u[1] = a0.y; fa.u[2] = a0.z; fa.u[3] = a0.w;
      fa.u[4] = a1.x; fa.u[5] = a1.y; fa.u[6] = a1.z; fa.u[7] = a1.w;
      // B fragment from packed global Wk (two b128 loads, contiguous stream)
      const uint4* bp = (const uint4*)(bt + (size_t)kt * 512);
      uint4 b0 = bp[0];
      uint4 b1 = bp[1];
      Frag fb;
      fb.u[0] = b0.x; fb.u[1] = b0.y; fb.u[2] = b0.z; fb.u[3] = b0.w;
      fb.u[4] = b1.x; fb.u[5] = b1.y; fb.u[6] = b1.z; fb.u[7] = b1.w;

      c = __builtin_amdgcn_wmma_f32_16x16x32_bf16(
              /*neg_a=*/false, fa.v, /*neg_b=*/false, fb.v,
              /*c_mod=*/(short)0, c, /*reuse_a=*/false, /*reuse_b=*/false);
    }
    // epilogue: element (r) of lane = kproj[s0 + r + 8*half][ht*16 + row]
    int   hcol = ht * 16 + row;
    float qv   = qproj[b * HH + hcol];
    float wv   = We[hcol];
#pragma unroll
    for (int r = 0; r < 8; ++r)
      rowacc[r] += fast_tanh(c[r] + qv) * wv;
  }

  // reduce over the 16 h-lanes of each half-wave; lane row==0 holds the sum
#pragma unroll
  for (int r = 0; r < 8; ++r) {
    float v = rowacc[r];
    v += __shfl_xor(v, 1, 32);
    v += __shfl_xor(v, 2, 32);
    v += __shfl_xor(v, 4, 32);
    v += __shfl_xor(v, 8, 32);
    if (row == 0)
      partial[wave][r + half * 8] = v;     // per-wave h-half partial
  }
  __syncthreads();
  // combine the two h-halves of each row region and write scores
  if (hhalf == 0 && lane < 16)
    scores[(size_t)b * SS + s0 + lane] = partial[wave][lane] + partial[wave + 4][lane];
}

// ---------------------------------------------------------------------------
// Kernel 4: masked softmax over S per batch; writes weights into d_out tail.
// ---------------------------------------------------------------------------
__global__ void softmax_kernel(const float* __restrict__ scores,
                               const int* __restrict__ mask,
                               float* __restrict__ wout) {
  const int b = blockIdx.x, t = threadIdx.x;
  __shared__ float red[256];
  float v[8];
  float m = -1e30f;
#pragma unroll
  for (int j = 0; j < 8; ++j) {
    int i = t + j * 256;
    float s = scores[(size_t)b * SS + i];
    if (mask[(size_t)b * SS + i] == 0) s = -1e9f;
    v[j] = s;
    m = fmaxf(m, s);
  }
  red[t] = m; __syncthreads();
  for (int o = 128; o > 0; o >>= 1) {
    if (t < o) red[t] = fmaxf(red[t], red[t + o]);
    __syncthreads();
  }
  m = red[0]; __syncthreads();
  float sum = 0.0f;
#pragma unroll
  for (int j = 0; j < 8; ++j) { v[j] = __expf(v[j] - m); sum += v[j]; }
  red[t] = sum; __syncthreads();
  for (int o = 128; o > 0; o >>= 1) {
    if (t < o) red[t] += red[t + o];
    __syncthreads();
  }
  float inv = 1.0f / red[0];
#pragma unroll
  for (int j = 0; j < 8; ++j)
    wout[(size_t)b * SS + t + j * 256] = v[j] * inv;
}

// ---------------------------------------------------------------------------
// Kernel 5a: partial contexts. Grid B*SCH WGs (4096 waves) to saturate HBM
// on the 512 MB values stream; each WG handles one (b, 256-s chunk).
// ---------------------------------------------------------------------------
__global__ void context_partial_kernel(const float* __restrict__ values,
                                       const float* __restrict__ w,
                                       float* __restrict__ part) {
  const int b  = blockIdx.x / SCH;
  const int ch = blockIdx.x % SCH;
  const int v0 = threadIdx.x * 4;
  const int sBeg = ch * (SS / SCH);
  const float* wr = w + (size_t)b * SS;
  const float* vb = values + (size_t)b * SS * VDV + v0;
  float ax = 0.f, ay = 0.f, az = 0.f, aw = 0.f;
  for (int s = sBeg; s < sBeg + SS / SCH; s += 4) {
#pragma unroll
    for (int u = 0; u < 4; ++u) {
      float ws  = wr[s + u];
      float4 vv = *(const float4*)(vb + (size_t)(s + u) * VDV);
      ax += ws * vv.x; ay += ws * vv.y; az += ws * vv.z; aw += ws * vv.w;
    }
  }
  float4 r; r.x = ax; r.y = ay; r.z = az; r.w = aw;
  *(float4*)(part + ((size_t)b * SCH + ch) * VDV + v0) = r;
}

// Kernel 5b: deterministic reduction of the SCH partials (no float atomics).
__global__ void context_reduce_kernel(const float* __restrict__ part,
                                      float* __restrict__ out) {
  int gid = blockIdx.x * 256 + threadIdx.x;   // 0 .. B*VD-1
  int b = gid >> 10;
  int v = gid & 1023;
  float s = 0.0f;
#pragma unroll
  for (int c = 0; c < SCH; ++c)
    s += part[((size_t)b * SCH + c) * VDV + v];
  out[(size_t)b * VDV + v] = s;
}

// ---------------------------------------------------------------------------
extern "C" void kernel_launch(void* const* d_in, const int* in_sizes, int n_in,
                              void* d_out, int out_size, void* d_ws, size_t ws_size,
                              hipStream_t stream) {
  (void)in_sizes; (void)n_in; (void)out_size; (void)ws_size;
  const float* query  = (const float*)d_in[0];
  const float* keys   = (const float*)d_in[1];
  const float* values = (const float*)d_in[2];
  const int*   mask   = (const int*)d_in[3];
  const float* Wq     = (const float*)d_in[4];
  const float* Wk     = (const float*)d_in[5];
  const float* We     = (const float*)d_in[6];
  float* out = (float*)d_out;

  // workspace: [wkbf bf16 2MB][qproj B*H f32][scores B*S f32][part B*SCH*VD f32]
  unsigned short* wkbf = (unsigned short*)d_ws;
  float* qproj  = (float*)((char*)d_ws + (size_t)HH * KDK * 2);
  float* scores = qproj + (size_t)BB * HH;
  float* part   = scores + (size_t)BB * SS;
  float* weights_out = out + (size_t)BB * VDV;   // tuple: (context, weights)

  pack_wk_bf16<<<(HH * KDK) / 256, 256, 0, stream>>>(Wk, wkbf);
  qproj_kernel<<<(BB * HH) / 256, 256, 0, stream>>>(query, Wq, qproj);
  fused_scores_kernel<<<BB * (SS / 64), 256, SMEM_BYTES, stream>>>(
      keys, qproj, We, wkbf, scores);
  softmax_kernel<<<BB, 256, 0, stream>>>(scores, mask, weights_out);
  context_partial_kernel<<<BB * SCH, 256, 0, stream>>>(values, weights_out, part);
  context_reduce_kernel<<<(BB * VDV) / 256, 256, 0, stream>>>(part, out);
}
